// UCIVanillaGRU_28140625723533
// MI455X (gfx1250) — matrix operand order repaired and checked by reading, not compiled, not run on hardware
//
#include <hip/hip_runtime.h>
#include <hip/hip_bf16.h>

typedef __bf16 bf16;
typedef __attribute__((ext_vector_type(16))) __bf16 v16bf;
typedef __attribute__((ext_vector_type(8)))  float  v8f;

#define T_STEPS 512
#define B_TOT   256
#define I_SZ    64
#define H_SZ    256
#define G_SZ    768   // 3H
#define BM      16    // batch rows per workgroup
#define NTILES  48    // G_SZ / 16
#define KB_H    8     // H / 32 k-steps (hidden GEMM)
#define KB_REG  4     // k-steps of w_hh kept in VGPRs
#define KB_LDS  4     // k-steps of w_hh kept in LDS
#define KB_I    2     // I / 32 k-steps (input GEMM)
#define WAVES   16
#define TPW     3     // N-tiles per wave (48 / 16)

// ---- workspace layout (bytes) ----
#define OFF_PHH 0
#define SZ_PHH  (NTILES * KB_H * 32 * 16 * 2)   // 393216 B: w_hh packed to WMMA-B lane layout
#define OFF_PIH (OFF_PHH + SZ_PHH)
#define SZ_PIH  (NTILES * KB_I * 32 * 16 * 2)   // 98304 B: w_ih packed
#define OFF_XBF (OFF_PIH + SZ_PIH)
#define SZ_XBF  (B_TOT * T_STEPS * I_SZ * 2)    // 16.8 MB: x in bf16

// WMMA-B lane layout (Kx16 tile): lane L holds column n = nt*16 + (L&15),
// 16 contiguous K elems starting at kb*32 + (L>>4)*16.
__global__ void pack_whh_kernel(const float* __restrict__ w, bf16* __restrict__ out) {
    int idx  = blockIdx.x * blockDim.x + threadIdx.x;   // NTILES*KB_H*32*16 threads
    int j    = idx & 15;
    int rec  = idx >> 4;
    int lane = rec & 31;
    int kb   = (rec >> 5) & 7;
    int nt   = rec >> 8;
    int n    = nt * 16 + (lane & 15);
    int k    = kb * 32 + (lane >> 4) * 16 + j;
    out[idx] = (bf16)w[n * H_SZ + k];
}

__global__ void pack_wih_kernel(const float* __restrict__ w, bf16* __restrict__ out) {
    int idx  = blockIdx.x * blockDim.x + threadIdx.x;   // NTILES*KB_I*32*16 threads
    int j    = idx & 15;
    int rec  = idx >> 4;
    int lane = rec & 31;
    int kb   = (rec >> 5) & 1;
    int nt   = rec >> 6;
    int n    = nt * 16 + (lane & 15);
    int k    = kb * 32 + (lane >> 4) * 16 + j;
    out[idx] = (bf16)w[n * I_SZ + k];
}

__global__ void cvt_x_kernel(const float* __restrict__ x, bf16* __restrict__ out, int n) {
    int i = blockIdx.x * blockDim.x + threadIdx.x;
    if (i < n) out[i] = (bf16)x[i];
}

union AV { v16bf v; uint4 q[2]; };

__device__ __forceinline__ float sigmoidf_(float x) { return 1.0f / (1.0f + __expf(-x)); }
__device__ __forceinline__ v8f vzero8() {
    v8f z = {0.f, 0.f, 0.f, 0.f, 0.f, 0.f, 0.f, 0.f};
    return z;
}

// Persistent GRU scan: 16 workgroups x 512 threads; each WG owns 16 batch rows
// for all 512 timesteps. w_hh k-steps 0..3 live in VGPRs, k-steps 4..7 in LDS.
__global__ __launch_bounds__(512) void gru_scan_kernel(
    const bf16*  __restrict__ xbf,
    const bf16*  __restrict__ phh,
    const bf16*  __restrict__ pih,
    const float* __restrict__ b_ih,
    const float* __restrict__ b_hh,
    const float* __restrict__ w_out,
    const float* __restrict__ b_out,
    float*       __restrict__ out)
{
    extern __shared__ char smem[];
    bf16*  Wl  = (bf16*)smem;                  // [48 tiles][4 kb][32 lanes][16] w_hh hi-K half (192 KB)
    float* Gh  = (float*)(smem + 196608);      // [16][768] hidden-side pre-activations (48 KB)
    float* Gi  = Gh + BM * G_SZ;               // [16][768] input-side pre-activations (48 KB)
    float* Hf  = Gi + BM * G_SZ;               // [16][256] h state, f32 (16 KB)
    bf16*  Hbf = (bf16*)(Hf + BM * H_SZ);      // [16][256] h state, bf16 WMMA-A source (8 KB)
    float* CB  = (float*)(Hbf + BM * H_SZ);    // [1281] folded biases + w_out + b_out (5 KB)

    const int tid  = threadIdx.x;
    const int wave = tid >> 5;
    const int lane = tid & 31;
    const int l16  = lane & 15;
    const int half = lane >> 4;
    const int b0   = blockIdx.x * BM;

    // ---- one-time setup ----
    // constants
    for (int i = tid; i < 2 * H_SZ; i += 512) CB[i] = b_ih[i] + b_hh[i];  // r,z bias sums
    for (int i = tid; i < H_SZ; i += 512) {
        CB[512  + i] = b_ih[2 * H_SZ + i];                                // n input-side bias
        CB[768  + i] = b_hh[2 * H_SZ + i];                                // n hidden-side bias
        CB[1024 + i] = w_out[i];
    }
    if (tid == 0) CB[1280] = b_out[0];
    // zero h state
    for (int i = tid; i < BM * H_SZ; i += 512) { Hf[i] = 0.0f; Hbf[i] = (bf16)0.0f; }
    // stage w_hh k-steps 4..7 into LDS: per tile nt, records (nt*8+4..7) are 4 KB contiguous
    for (int i = tid; i < NTILES * 256; i += 512) {       // 12288 uint4 = 192 KB
        int nt  = i >> 8;
        int rem = i & 255;
        ((uint4*)Wl)[nt * 256 + rem] =
            *((const uint4*)(phh + (size_t)(nt * KB_H + KB_REG) * 512) + rem);
    }

    // register-resident w_hh k-steps 0..3: 3 N-tiles x 4 k-steps per wave (96 VGPRs)
    v16bf Bhh[TPW][KB_REG];
#pragma unroll
    for (int it = 0; it < TPW; ++it) {
        const int nt = wave * TPW + it;
#pragma unroll
        for (int kb = 0; kb < KB_REG; ++kb)
            Bhh[it][kb] = *(const v16bf*)(phh + ((size_t)(nt * KB_H + kb) * 32 + lane) * 16);
    }
    __syncthreads();

    const bf16* xrow = xbf + ((size_t)(b0 + l16) * T_STEPS) * I_SZ;

    for (int t = 0; t < T_STEPS; ++t) {
        v8f acc[TPW];
#pragma unroll
        for (int it = 0; it < TPW; ++it) acc[it] = vzero8();

        // ---- input-side GEMM: x_t @ w_ih^T (K = 64) ----
#pragma unroll
        for (int kb = 0; kb < KB_I; ++kb) {
            AV a;
            const bf16* p = xrow + t * I_SZ + kb * 32 + half * 16;
            a.q[0] = *(const uint4*)(p);
            a.q[1] = *(const uint4*)(p + 8);
#pragma unroll
            for (int it = 0; it < TPW; ++it) {
                v16bf bw = *(const v16bf*)(pih + ((size_t)((wave * TPW + it) * KB_I + kb) * 32 + lane) * 16);
                acc[it] = __builtin_amdgcn_wmma_f32_16x16x32_bf16(
                    false, a.v, false, bw, (short)0, acc[it], false, false);
            }
        }
        if (t + 1 < T_STEPS) __builtin_prefetch(xrow + (t + 1) * I_SZ, 0, 1);

        // dump input-side partials (uniform across all tiles; n-gate needs them separate)
#pragma unroll
        for (int it = 0; it < TPW; ++it) {
            const int nt = wave * TPW + it;
#pragma unroll
            for (int r = 0; r < 8; ++r)
                Gi[(half * 8 + r) * G_SZ + nt * 16 + l16] = acc[it][r];
            acc[it] = vzero8();
        }

        // ---- hidden-side GEMM: h @ w_hh^T (K = 256) ----
        // k-steps 0..3: weights from VGPRs
#pragma unroll
        for (int kb = 0; kb < KB_REG; ++kb) {
            AV a;
            const bf16* hp = Hbf + l16 * H_SZ + kb * 32 + half * 16;
            a.q[0] = *(const uint4*)(hp);
            a.q[1] = *(const uint4*)(hp + 8);
#pragma unroll
            for (int it = 0; it < TPW; ++it)
                acc[it] = __builtin_amdgcn_wmma_f32_16x16x32_bf16(
                    false, a.v, false, Bhh[it][kb], (short)0, acc[it], false, false);
        }
        // k-steps 4..7: weights from LDS
#pragma unroll
        for (int kb = 0; kb < KB_LDS; ++kb) {
            AV a;
            const bf16* hp = Hbf + l16 * H_SZ + (KB_REG + kb) * 32 + half * 16;
            a.q[0] = *(const uint4*)(hp);
            a.q[1] = *(const uint4*)(hp + 8);
#pragma unroll
            for (int it = 0; it < TPW; ++it) {
                const int nt = wave * TPW + it;
                v16bf bw = *(const v16bf*)(Wl + ((size_t)(nt * KB_LDS + kb) * 32 + lane) * 16);
                acc[it] = __builtin_amdgcn_wmma_f32_16x16x32_bf16(
                    false, a.v, false, bw, (short)0, acc[it], false, false);
            }
        }

        // dump hidden-side pre-activations
#pragma unroll
        for (int it = 0; it < TPW; ++it) {
            const int nt = wave * TPW + it;
#pragma unroll
            for (int r = 0; r < 8; ++r)
                Gh[(half * 8 + r) * G_SZ + nt * 16 + l16] = acc[it][r];
        }
        __syncthreads();

        // ---- gate combine: wave m owns batch row m, 8 hidden units per lane ----
        {
            const int m  = wave;
            const int j0 = lane * 8;
            float oacc = 0.0f;
#pragma unroll
            for (int jj = 0; jj < 8; ++jj) {
                const int j = j0 + jj;
                const float r  = sigmoidf_(Gi[m * G_SZ + j] + Gh[m * G_SZ + j] + CB[j]);
                const float z  = sigmoidf_(Gi[m * G_SZ + H_SZ + j] + Gh[m * G_SZ + H_SZ + j] + CB[H_SZ + j]);
                const float nn = tanhf(Gi[m * G_SZ + 2 * H_SZ + j] + CB[512 + j] +
                                       r * (Gh[m * G_SZ + 2 * H_SZ + j] + CB[768 + j]));
                const float hp = Hf[m * H_SZ + j];
                const float hn = (1.0f - z) * nn + z * hp;
                Hf[m * H_SZ + j]  = hn;
                Hbf[m * H_SZ + j] = (bf16)hn;
                oacc += hn * CB[1024 + j];
            }
#pragma unroll
            for (int off = 16; off > 0; off >>= 1) oacc += __shfl_down(oacc, off, 32);
            if (lane == 0) out[t * B_TOT + b0 + m] = oacc + CB[1280];
        }
        __syncthreads();
    }
}

extern "C" void kernel_launch(void* const* d_in, const int* in_sizes, int n_in,
                              void* d_out, int out_size, void* d_ws, size_t ws_size,
                              hipStream_t stream) {
    const float* x     = (const float*)d_in[0];
    const float* w_ih  = (const float*)d_in[1];
    const float* w_hh  = (const float*)d_in[2];
    const float* b_ih  = (const float*)d_in[3];
    const float* b_hh  = (const float*)d_in[4];
    const float* w_out = (const float*)d_in[5];
    const float* b_out = (const float*)d_in[6];
    float* out = (float*)d_out;

    char* ws = (char*)d_ws;
    bf16* phh = (bf16*)(ws + OFF_PHH);
    bf16* pih = (bf16*)(ws + OFF_PIH);
    bf16* xbf = (bf16*)(ws + OFF_XBF);

    pack_whh_kernel<<<(NTILES * KB_H * 32 * 16) / 256, 256, 0, stream>>>(w_hh, phh);
    pack_wih_kernel<<<(NTILES * KB_I * 32 * 16) / 256, 256, 0, stream>>>(w_ih, pih);
    const int nx = B_TOT * T_STEPS * I_SZ;
    cvt_x_kernel<<<(nx + 255) / 256, 256, 0, stream>>>(x, xbf, nx);

    const size_t smem_bytes =
        196608                                     /* Wl: w_hh hi-K half */
        + (size_t)(2 * BM * G_SZ) * sizeof(float)  /* Gh + Gi */
        + (size_t)(BM * H_SZ) * sizeof(float)      /* Hf */
        + (size_t)(BM * H_SZ) * sizeof(bf16)       /* Hbf */
        + (size_t)1281 * sizeof(float);            /* CB */
    hipFuncSetAttribute(reinterpret_cast<const void*>(gru_scan_kernel),
                        hipFuncAttributeMaxDynamicSharedMemorySize, (int)smem_bytes);
    gru_scan_kernel<<<B_TOT / BM, 512, smem_bytes, stream>>>(
        xbf, phh, pih, b_ih, b_hh, w_out, b_out, out);
}
